// GATLayer_32229434589224
// MI455X (gfx1250) — matrix operand order repaired and checked by reading, not compiled, not run on hardware
//
#include <hip/hip_runtime.h>
#include <hip/hip_bf16.h>
#include <stdint.h>

// GAT layer: N=8192 nodes, IN=512 -> OUT=256, masked softmax attention.
#define NN      8192
#define IN_DIM  512
#define OUT_DIM 256
#define ALPHA   0.2f
#define NEG_INF (-9.0e15f)

typedef __attribute__((ext_vector_type(16))) _Float16 v16h;
typedef __attribute__((ext_vector_type(8)))  _Float16 h8;
typedef __attribute__((ext_vector_type(8)))  float    v8f;
typedef __attribute__((ext_vector_type(4)))  float    f4;

// ---------------------------------------------------------------------------
// 1) fp32 -> fp16 conversion of node features (row-major, 8192x512)
// ---------------------------------------------------------------------------
__global__ void cvt_x_kernel(const float* __restrict__ X, _Float16* __restrict__ Xh) {
    int idx = blockIdx.x * 256 + threadIdx.x;      // 0 .. N*IN-1
    Xh[idx] = (_Float16)X[idx];
}

// ---------------------------------------------------------------------------
// 2) W (512x256, row-major) -> WT fp16 (256x512): WT[n][k] = W[k][n]
//    Writes coalesced; strided reads stay in L2 (W is only 512 KB).
// ---------------------------------------------------------------------------
__global__ void cvt_w_kernel(const float* __restrict__ W, _Float16* __restrict__ WT) {
    int idx = blockIdx.x * 256 + threadIdx.x;      // 0 .. OUT*IN-1
    int n = idx >> 9;                              // / IN_DIM
    int k = idx & (IN_DIM - 1);
    WT[idx] = (_Float16)W[k * OUT_DIM + n];
}

// ---------------------------------------------------------------------------
// 3) h = X @ W + b via v_wmma_f32_16x16x32_f16, one 16x16 tile per wave.
//    Result stored TRANSPOSED as fp16: hT[n][i] (256 x 8192) so that the
//    attention GEMM's B fragments are contiguous b128 loads.
//    A layout (16-bit A 16x32): lane row = lane&15; K-halves split by lane>=16.
//    B layout (16-bit B 32x16): lane col = lane&15; K 0..7,16..23 / 8..15,24..31 by half.
// ---------------------------------------------------------------------------
__global__ void gemm_h_kernel(const _Float16* __restrict__ Xh,
                              const _Float16* __restrict__ WT,
                              const float* __restrict__ bias,
                              _Float16* __restrict__ hT) {
    const int lane = threadIdx.x;          // 0..31 (wave32)
    const int hi   = lane >> 4;            // lo/hi half of wave
    const int lm   = lane & 15;
    const int i0   = blockIdx.x * 16;      // node tile
    const int n0   = blockIdx.y * 16;      // feature tile
    const int n    = n0 + lm;

    const _Float16* arow = Xh + (size_t)(i0 + lm) * IN_DIM;
    const _Float16* brow = WT + (size_t)n * IN_DIM;

    v8f acc = {};
    #pragma unroll 4
    for (int kb = 0; kb < IN_DIM; kb += 32) {
        h8 a0 = *(const h8*)(arow + kb + hi * 8);
        h8 a1 = *(const h8*)(arow + kb + 16 + hi * 8);
        h8 b0 = *(const h8*)(brow + kb + hi * 8);
        h8 b1 = *(const h8*)(brow + kb + 16 + hi * 8);
        v16h A = __builtin_shufflevector(a0, a1, 0,1,2,3,4,5,6,7,8,9,10,11,12,13,14,15);
        v16h B = __builtin_shufflevector(b0, b1, 0,1,2,3,4,5,6,7,8,9,10,11,12,13,14,15);
        acc = __builtin_amdgcn_wmma_f32_16x16x32_f16(false, A, false, B,
                                                     (short)0, acc, false, false);
    }
    // C/D layout: element v -> row M = hi*8 + v, col N = lm.
    const float bv = bias[n];
    h8 outv;
    #pragma unroll
    for (int v = 0; v < 8; ++v) outv[v] = (_Float16)(acc[v] + bv);
    // hT[n][i0 + hi*8 + 0..7] : contiguous 16B store per lane
    *(h8*)(hT + (size_t)n * NN + i0 + hi * 8) = outv;
}

// ---------------------------------------------------------------------------
// 4) e_l[i] = sum_n h[i][n]*a[n] (+ a_bias folded in);
//    e_r[i] = sum_n h[i][n]*a[OUT+n]
//    hT is feature-major so threads (consecutive i) read coalesced.
// ---------------------------------------------------------------------------
__global__ void el_er_kernel(const _Float16* __restrict__ hT,
                             const float* __restrict__ a,
                             const float* __restrict__ a_bias,
                             float* __restrict__ el, float* __restrict__ er) {
    int i = blockIdx.x * 256 + threadIdx.x;        // node id
    float sl = 0.f, sr = 0.f;
    for (int n = 0; n < OUT_DIM; ++n) {
        float v = (float)hT[(size_t)n * NN + i];
        sl += v * a[n];
        sr += v * a[OUT_DIM + n];
    }
    el[i] = sl + a_bias[0];                        // fold bias into e_l
    er[i] = sr;
}

// ---------------------------------------------------------------------------
// 5) Per-row softmax stats + adjacency bit-pack.
//    One block (256 thr = 8 waves) per row. Pass 1: read adj row once,
//    __ballot-pack into 1 dword / 32 cols (written to global abits),
//    track masked leaky-relu max. Pass 2: sum of exp(score - max) using the
//    per-thread bit record (no adj re-read). adj is touched exactly once.
// ---------------------------------------------------------------------------
__global__ void stats_kernel(const int* __restrict__ adj,
                             const float* __restrict__ el,
                             const float* __restrict__ er,
                             unsigned* __restrict__ abits,
                             float* __restrict__ rmax, float* __restrict__ rsum) {
    __shared__ float smax[8];
    __shared__ float ssum[8];

    const int i    = blockIdx.x;
    const int t    = threadIdx.x;
    const int wave = t >> 5;
    const int lane = t & 31;
    const float eli = el[i];                       // a_bias already folded in
    const int* arow = adj + (size_t)i * NN;

    unsigned mybits = 0;
    float lmax = NEG_INF;
    for (int k = 0; k < 32; ++k) {
        int j = k * 256 + wave * 32 + lane;        // wave covers 32 consecutive j
        int av = arow[j];
        unsigned m = (unsigned)__ballot(av > 0);
        if (lane == 0) abits[(size_t)i * (NN / 32) + (j >> 5)] = m;
        if (av > 0) {
            mybits |= 1u << k;
            float s  = eli + er[j];
            float lr = s > 0.f ? s : ALPHA * s;
            lmax = fmaxf(lmax, lr);
        }
    }
    #pragma unroll
    for (int off = 16; off; off >>= 1) lmax = fmaxf(lmax, __shfl_xor(lmax, off, 32));
    if (lane == 0) smax[wave] = lmax;
    __syncthreads();
    float rowmax = smax[0];
    #pragma unroll
    for (int w = 1; w < 8; ++w) rowmax = fmaxf(rowmax, smax[w]);

    float lsum = 0.f;
    for (int k = 0; k < 32; ++k) {
        if ((mybits >> k) & 1u) {
            int j = k * 256 + wave * 32 + lane;
            float s  = eli + er[j];
            float lr = s > 0.f ? s : ALPHA * s;
            lsum += __expf(lr - rowmax);
        }
    }
    #pragma unroll
    for (int off = 16; off; off >>= 1) lsum += __shfl_xor(lsum, off, 32);
    if (lane == 0) ssum[wave] = lsum;
    __syncthreads();
    if (t == 0) {
        float tot = 0.f;
        #pragma unroll
        for (int w = 0; w < 8; ++w) tot += ssum[w];
        rmax[i] = rowmax;
        rsum[i] = tot;
    }
}

// ---------------------------------------------------------------------------
// 6) h' = softmax(scores) @ h, flash-style: attention tile (16x32 fp16) is
//    rebuilt in registers from el/er/bits/rowmax each K step and fed to WMMA.
//    One wave -> 16 rows x 128 output features (8 accumulators = 64 VGPRs):
//    each A-build (16 v_exp_f32) is amortized over 8 WMMAs, and the softmax
//    1/rsum normalization is folded into the fp16 weights (A rows are
//    per-lane, so 1/rsum[irow] is a per-lane scalar) -> no epilogue divide.
// ---------------------------------------------------------------------------
__global__ void flash_kernel(const _Float16* __restrict__ hT,
                             const unsigned* __restrict__ abits,
                             const float* __restrict__ el,
                             const float* __restrict__ er,
                             const float* __restrict__ rmax,
                             const float* __restrict__ rsum,
                             float* __restrict__ out) {
    const int lane = threadIdx.x;
    const int hi   = lane >> 4;
    const int lm   = lane & 15;
    const int i0   = blockIdx.x * 16;          // node tile
    const int nb   = blockIdx.y * 128;         // output-feature tile

    const int irow    = i0 + lm;               // A-matrix row owned by this lane
    const float eli   = el[irow];              // a_bias already folded in
    const float mi    = rmax[irow];
    const float wscal = 1.0f / rsum[irow];     // softmax normalization, folded in
    const unsigned* bitsrow = abits + (size_t)irow * (NN / 32);

    v8f acc[8] = {{}, {}, {}, {}, {}, {}, {}, {}};

    for (int jb = 0; jb < NN; jb += 32) {
        unsigned m = bitsrow[jb >> 5];         // bits for cols jb .. jb+31
        const int lo0 = jb + hi * 8;
        const int lo1 = jb + 16 + hi * 8;
        f4 e0 = *(const f4*)(er + lo0);
        f4 e1 = *(const f4*)(er + lo0 + 4);
        f4 e2 = *(const f4*)(er + lo1);
        f4 e3 = *(const f4*)(er + lo1 + 4);

        v16h A;
        #pragma unroll
        for (int q = 0; q < 8; ++q) {
            float erv = (q < 4) ? e0[q] : e1[q - 4];
            float s   = eli + erv;
            float lr  = s > 0.f ? s : ALPHA * s;
            float w   = ((m >> (hi * 8 + q)) & 1u) ? __expf(lr - mi) * wscal : 0.f;
            A[q] = (_Float16)w;                // K = jb + hi*8 + q

            float erv2 = (q < 4) ? e2[q] : e3[q - 4];
            float s2   = eli + erv2;
            float lr2  = s2 > 0.f ? s2 : ALPHA * s2;
            float w2   = ((m >> (16 + hi * 8 + q)) & 1u) ? __expf(lr2 - mi) * wscal : 0.f;
            A[q + 8] = (_Float16)w2;           // K = jb + 16 + hi*8 + q
        }

        const _Float16* hb = hT + jb + hi * 8;
        #pragma unroll
        for (int tt = 0; tt < 8; ++tt) {
            const _Float16* p = hb + (size_t)(nb + tt * 16 + lm) * NN;
            h8 b0 = *(const h8*)(p);
            h8 b1 = *(const h8*)(p + 16);
            v16h B = __builtin_shufflevector(b0, b1, 0,1,2,3,4,5,6,7,8,9,10,11,12,13,14,15);
            acc[tt] = __builtin_amdgcn_wmma_f32_16x16x32_f16(false, A, false, B,
                                                             (short)0, acc[tt],
                                                             false, false);
        }
    }

    // C/D rows for this lane: i0 + hi*8 + v; already normalized via A weights.
    #pragma unroll
    for (int tt = 0; tt < 8; ++tt) {
        #pragma unroll
        for (int v = 0; v < 8; ++v) {
            out[(size_t)(i0 + hi * 8 + v) * OUT_DIM + nb + tt * 16 + lm] = acc[tt][v];
        }
    }
}

// ---------------------------------------------------------------------------
// Workspace layout (bytes):
//   Xh    fp16  N*IN        @ 0         ( 8 MB)
//   WT    fp16  OUT*IN      @ 8388608   (256 KB)
//   hT    fp16  OUT*N       @ 8650752   ( 4 MB)
//   el    f32   N           @ 12845056
//   er    f32   N           @ 12877824
//   rmax  f32   N           @ 12910592
//   rsum  f32   N           @ 12943360
//   abits u32   N*N/32      @ 12976128  ( 8 MB)   total ~20.4 MB
// ---------------------------------------------------------------------------
extern "C" void kernel_launch(void* const* d_in, const int* in_sizes, int n_in,
                              void* d_out, int out_size, void* d_ws, size_t ws_size,
                              hipStream_t stream) {
    const int*   adj   = (const int*)  d_in[0];
    const float* X     = (const float*)d_in[1];
    const float* W     = (const float*)d_in[2];
    const float* bvec  = (const float*)d_in[3];
    const float* avec  = (const float*)d_in[4];
    const float* abias = (const float*)d_in[5];
    float* out = (float*)d_out;

    char* ws = (char*)d_ws;
    _Float16* Xh   = (_Float16*)(ws);
    _Float16* WT   = (_Float16*)(ws + 8388608);
    _Float16* hT   = (_Float16*)(ws + 8650752);
    float*    el   = (float*)   (ws + 12845056);
    float*    er   = (float*)   (ws + 12877824);
    float*    rmax = (float*)   (ws + 12910592);
    float*    rsum = (float*)   (ws + 12943360);
    unsigned* abit = (unsigned*)(ws + 12976128);

    cvt_x_kernel <<<(NN * IN_DIM) / 256, 256, 0, stream>>>(X, Xh);
    cvt_w_kernel <<<(IN_DIM * OUT_DIM) / 256, 256, 0, stream>>>(W, WT);
    gemm_h_kernel<<<dim3(NN / 16, OUT_DIM / 16), 32, 0, stream>>>(Xh, WT, bvec, hT);
    el_er_kernel <<<NN / 256, 256, 0, stream>>>(hT, avec, abias, el, er);
    stats_kernel <<<NN, 256, 0, stream>>>(adj, el, er, abit, rmax, rsum);
    flash_kernel <<<dim3(NN / 16, OUT_DIM / 128), 32, 0, stream>>>(hT, abit, el, er,
                                                                   rmax, rsum, out);
}
